// PartialFC_V2_SubCenter_57750130262074
// MI455X (gfx1250) — matrix operand order repaired and checked by reading, not compiled, not run on hardware
//
#include <hip/hip_runtime.h>
#include <hip/hip_bf16.h>
#include <math.h>

typedef _Float16 v16h __attribute__((ext_vector_type(16)));
typedef _Float16 v8h  __attribute__((ext_vector_type(8)));
typedef float    v8f  __attribute__((ext_vector_type(8)));

namespace {
constexpr int   kD       = 512;      // feature dim
constexpr int   kB       = 128;      // batch
constexpr int   kC       = 100000;   // classes
constexpr int   kSub     = 3;        // sub-centers
constexpr int   kCK      = kC * kSub;        // 300000 weight rows
constexpr int   kTile    = 48;               // sub-center rows per WG (16 classes)
constexpr int   kWStride = 536;              // LDS row stride in f16 (512 + 24 pad; %64dw == 12)
constexpr float kScale   = 64.0f;
constexpr float kCosM    = 0.87758256189037271f;   // cos(0.5)
constexpr float kSinM    = 0.47942553860420301f;   // sin(0.5)
constexpr float kThresh  = -0.87758256189037271f;  // cos(pi - 0.5)
constexpr float kSinMM   = 0.23971276930210150f;   // sin(pi - 0.5) * 0.5
}

// ---------------------------------------------------------------- kernel 1 --
// Normalize embeddings (fp32) and emit f16 rows for the WMMA A operand.
__global__ __launch_bounds__(256) void pfc_norm(const float* __restrict__ E,
                                                _Float16* __restrict__ neh) {
  __shared__ float red[256];
  const int b = blockIdx.x, tid = threadIdx.x;
  const float x0 = E[b * kD + tid];
  const float x1 = E[b * kD + tid + 256];
  red[tid] = x0 * x0 + x1 * x1;
  __syncthreads();
  #pragma unroll
  for (int s = 128; s > 0; s >>= 1) {
    if (tid < s) red[tid] += red[tid + s];
    __syncthreads();
  }
  const float rn = rsqrtf(red[0]);
  neh[b * kD + tid]       = (_Float16)(x0 * rn);
  neh[b * kD + tid + 256] = (_Float16)(x1 * rn);
}

// ---------------------------------------------------------------- kernel 2 --
// Per WG: 48-row weight tile. Stage fp32 -> f16 LDS + fp32 row norms, then
// 8 waves x (M=16 tile) x (3 N tiles) x (K=512 in steps of 32) f16 WMMA.
// Epilogue: scale by 1/||w||, clamp, max over 3 sub-centers, store [128,16].
__global__ __launch_bounds__(256) void pfc_gemm(const float* __restrict__ Wt,
                                                const _Float16* __restrict__ neh,
                                                float* __restrict__ logits) {
  __shared__ __align__(16) union {
    struct {
      _Float16 w[kTile * kWStride];  // 51456 B, f16 weight tile
      float    rn[kTile];            // 1/||w_row||
    } s;
    float ob[kB * kTile];            // 24576 B epilogue buffer (aliases w only)
  } sm;

  const int tid  = threadIdx.x;
  const int wv   = tid >> 5;         // wave 0..7
  const int lane = tid & 31;
  const int wg   = blockIdx.x;       // 0..6249

  // ---- stage weight tile: 6 rows per wave, 16 fp32 per lane per row ----
  #pragma unroll
  for (int rr = 0; rr < 6; ++rr) {
    const int r = wv * 6 + rr;                       // 0..47
    const size_t R = (size_t)wg * kTile + r;         // global weight row
    const float4* wp = (const float4*)(Wt + R * kD + lane * 16);
    const float4 f0 = wp[0], f1 = wp[1], f2 = wp[2], f3 = wp[3];
    float ss = f0.x*f0.x + f0.y*f0.y + f0.z*f0.z + f0.w*f0.w
             + f1.x*f1.x + f1.y*f1.y + f1.z*f1.z + f1.w*f1.w
             + f2.x*f2.x + f2.y*f2.y + f2.z*f2.z + f2.w*f2.w
             + f3.x*f3.x + f3.y*f3.y + f3.z*f3.z + f3.w*f3.w;
    const v8h h0 = { (_Float16)f0.x, (_Float16)f0.y, (_Float16)f0.z, (_Float16)f0.w,
                     (_Float16)f1.x, (_Float16)f1.y, (_Float16)f1.z, (_Float16)f1.w };
    const v8h h1 = { (_Float16)f2.x, (_Float16)f2.y, (_Float16)f2.z, (_Float16)f2.w,
                     (_Float16)f3.x, (_Float16)f3.y, (_Float16)f3.z, (_Float16)f3.w };
    *(v8h*)&sm.s.w[r * kWStride + lane * 16]     = h0;
    *(v8h*)&sm.s.w[r * kWStride + lane * 16 + 8] = h1;
    #pragma unroll
    for (int off = 16; off > 0; off >>= 1) ss += __shfl_xor(ss, off);
    if (lane == 0) sm.s.rn[r] = rsqrtf(ss);
  }
  __syncthreads();

  // ---- WMMA main loop ----
  const int l = lane & 15;           // N column / M row within 16
  const int h = lane >> 4;           // lane half (selects K sub-range)
  const _Float16* Abase = neh + ((wv << 4) + l) * kD;   // A row = 16*wave + l

  v8f acc[3] = {};
  #pragma unroll 4
  for (int kk = 0; kk < kD; kk += 32) {
    // A 16x32 f16 frag (ISA layout): lane holds K = kk+8h..+7 and kk+16+8h..+7
    const _Float16* ap = Abase + kk + (h << 3);
    const v8h alo = *(const v8h*)ap;
    const v8h ahi = *(const v8h*)(ap + 16);
    const v16h A = __builtin_shufflevector(alo, ahi,
        0,1,2,3,4,5,6,7,8,9,10,11,12,13,14,15);
    #pragma unroll
    for (int nt = 0; nt < 3; ++nt) {
      // B 32x16 frag: col j = l  ->  weight row nt*16+l, K = kk+8h..+7 and kk+16+8h..+7
      const _Float16* bp = &sm.s.w[(nt * 16 + l) * kWStride + kk + (h << 3)];
      const v8h blo = *(const v8h*)bp;
      const v8h bhi = *(const v8h*)(bp + 16);
      const v16h Bm = __builtin_shufflevector(blo, bhi,
          0,1,2,3,4,5,6,7,8,9,10,11,12,13,14,15);
      acc[nt] = __builtin_amdgcn_wmma_f32_16x16x32_f16(
          false, A, false, Bm, (short)0, acc[nt], false, false);
    }
  }
  __syncthreads();   // all B-frag LDS reads done before ob writes

  // ---- epilogue: scale by 1/||w||, clamp, spill to LDS ----
  float rns[3];
  #pragma unroll
  for (int nt = 0; nt < 3; ++nt) rns[nt] = sm.s.rn[nt * 16 + l];  // disjoint from ob
  #pragma unroll
  for (int nt = 0; nt < 3; ++nt) {
    #pragma unroll
    for (int g = 0; g < 8; ++g) {
      // C layout: VGPR g, lanes0-15 -> M=g, lanes16-31 -> M=8+g; N = l
      float v = acc[nt][g] * rns[nt];
      v = fminf(1.0f, fmaxf(-1.0f, v));
      sm.ob[((wv << 4) + g + (h << 3)) * kTile + nt * 16 + l] = v;
    }
  }
  __syncthreads();

  // ---- max over K=3 sub-centers, store 128x16 classes ----
  for (int i = tid; i < kB * 16; i += 256) {
    const int b = i >> 4, c = i & 15;
    const float* p = &sm.ob[b * kTile + c * 3];
    const float mx = fmaxf(p[0], fmaxf(p[1], p[2]));
    logits[(size_t)b * kC + (size_t)wg * 16 + c] = mx;
  }
}

// ---------------------------------------------------------------- kernel 3 --
__device__ __forceinline__ float arc_margin(float t) {
  const float sin_t = sqrtf(fmaxf(1.0f - t * t, 0.0f));
  const float ctm = t * kCosM - sin_t * kSinM;
  return (t > kThresh) ? ctm : (t - kSinMM);
}

// One block per batch row: margin on target, deterministic max + sum-exp.
__global__ __launch_bounds__(256) void pfc_rowloss(const float* __restrict__ logits,
                                                   const int* __restrict__ labels,
                                                   float* __restrict__ rloss) {
  __shared__ float red[256];
  const int b = blockIdx.x, tid = threadIdx.x;
  const int lab = labels[b];
  const float* row = logits + (size_t)b * kC;
  const float adj = arc_margin(row[lab]);   // uniform broadcast read

  float mx = -4.0f;                         // logits in [-1,1], margin >= -1.24
  for (int c = tid; c < kC; c += 256) {
    const float x = (c == lab) ? adj : row[c];
    mx = fmaxf(mx, x);
  }
  red[tid] = mx;
  __syncthreads();
  #pragma unroll
  for (int s = 128; s > 0; s >>= 1) {
    if (tid < s) red[tid] = fmaxf(red[tid], red[tid + s]);
    __syncthreads();
  }
  const float m = red[0];
  __syncthreads();

  float sum = 0.0f;
  for (int c = tid; c < kC; c += 256) {
    const float x = (c == lab) ? adj : row[c];
    sum += __expf(kScale * (x - m));
  }
  red[tid] = sum;
  __syncthreads();
  #pragma unroll
  for (int s = 128; s > 0; s >>= 1) {
    if (tid < s) red[tid] += red[tid + s];
    __syncthreads();
  }
  if (tid == 0) rloss[b] = -(kScale * (adj - m) - logf(red[0]));
}

// ---------------------------------------------------------------- kernel 4 --
__global__ __launch_bounds__(128) void pfc_final(const float* __restrict__ rloss,
                                                 float* __restrict__ out) {
  __shared__ float red[128];
  const int tid = threadIdx.x;
  red[tid] = rloss[tid];
  __syncthreads();
  #pragma unroll
  for (int s = 64; s > 0; s >>= 1) {
    if (tid < s) red[tid] += red[tid + s];
    __syncthreads();
  }
  if (tid == 0) out[0] = red[0] * (1.0f / 128.0f);
}

// --------------------------------------------------------------------------
extern "C" void kernel_launch(void* const* d_in, const int* in_sizes, int n_in,
                              void* d_out, int out_size, void* d_ws, size_t ws_size,
                              hipStream_t stream) {
  const float* emb    = (const float*)d_in[0];   // [128, 512] fp32
  const int*   labels = (const int*)d_in[1];     // [128] (int per harness convention)
  const float* weight = (const float*)d_in[2];   // [300000, 512] fp32
  float* out = (float*)d_out;

  // workspace layout
  const size_t logitsBytes = (size_t)kB * kC * sizeof(float);      // 51.2 MB (L2-resident)
  float*    logitsW = (float*)d_ws;
  _Float16* neh     = (_Float16*)((char*)d_ws + logitsBytes);      // 128 KB
  float*    rloss   = (float*)((char*)d_ws + logitsBytes + (size_t)kB * kD * sizeof(_Float16));

  pfc_norm<<<kB, 256, 0, stream>>>(emb, neh);
  pfc_gemm<<<kCK / kTile, 256, 0, stream>>>(weight, neh, logitsW);  // 6250 WGs
  pfc_rowloss<<<kB, 256, 0, stream>>>(logitsW, labels, rloss);
  pfc_final<<<1, 128, 0, stream>>>(rloss, out);
}
